// BaselineModel_55018531061929
// MI455X (gfx1250) — matrix operand order repaired and checked by reading, not compiled, not run on hardware
//
#include <hip/hip_runtime.h>
#include <hip/hip_bf16.h>

// ---------------------------------------------------------------------------
// Model constants (from reference): B=2, L=2048, D=1024, H=8, NL=2, HD=128
// ---------------------------------------------------------------------------
#define BB   2
#define LL   2048
#define DD   1024
#define HD   128
#define NLAY 2
#define MTOK (BB * LL)               // 4096 rows
#define EPS  1e-8f

typedef __bf16 bf16;
typedef __attribute__((ext_vector_type(16))) __bf16 v16bf;
typedef __attribute__((ext_vector_type(8)))  __bf16 v8bf;
typedef __attribute__((ext_vector_type(8)))  float  v8f;
typedef __attribute__((ext_vector_type(4)))  unsigned int v4u;
typedef __attribute__((ext_vector_type(8)))  int v8i;
typedef __attribute__((ext_vector_type(4)))  int v4i;

// ---------------------------------------------------------------------------
// Tensor Data Mover: async 2D tile DMA global -> LDS (CDNA5 TENSOR_LOAD_TO_LDS)
// D# layout per CDNA5 ISA.
//   group0: [1:0]=count=1 | [63:32]=lds_addr | [120:64]=global_addr | [127:126]=type(2)
//   group1: [17:16]=data_size(1=2B) | [79:48]=tensor_dim0 | [111:80]=tensor_dim1 |
//           [127:112]=tile_dim0 | [143:128]=tile_dim1 | [207:160]=tensor_dim0_stride
//   group2/3: zero (2D tensor)
// dims/strides in data_size units. Issued by one wave; tracked with TENSORcnt.
// This toolchain exposes the 6-arg builtin (extra int32x8 group before cpol).
// ---------------------------------------------------------------------------
__device__ __forceinline__ void tdm_load_2d(unsigned lds_off, const void* gptr,
                                            unsigned td0, unsigned td1,
                                            unsigned long long stride0,
                                            unsigned tile0, unsigned tile1) {
  const unsigned long long ga = (unsigned long long)gptr;
  v4u g0;
  g0[0] = 1u;                                        // count=1 (valid), user mode
  g0[1] = lds_off;                                   // LDS byte address
  g0[2] = (unsigned)(ga & 0xFFFFFFFFull);            // global_addr[31:0]
  g0[3] = (unsigned)((ga >> 32) & 0x1FFFFFFull) | (2u << 30);  // addr[56:32]|type=2
  v8i g1;
  g1[0] = (int)(1u << 16);                           // workgroup_mask=0, data_size=2B
  g1[1] = (int)((td0 & 0xFFFFu) << 16);              // tensor_dim0[15:0]
  g1[2] = (int)(((td0 >> 16) & 0xFFFFu) | ((td1 & 0xFFFFu) << 16));
  g1[3] = (int)(((td1 >> 16) & 0xFFFFu) | ((tile0 & 0xFFFFu) << 16));
  g1[4] = (int)(tile1 & 0xFFFFu);                    // tile_dim1, tile_dim2=0
  g1[5] = (int)(stride0 & 0xFFFFFFFFull);            // tensor_dim0_stride[31:0]
  g1[6] = (int)((stride0 >> 32) & 0xFFFFull);        // stride[47:32], dim1_stride=0
  g1[7] = 0;
  v4i g2 = {0, 0, 0, 0};
  v4i g3 = {0, 0, 0, 0};
  v8i g4 = {0, 0, 0, 0, 0, 0, 0, 0};
  __builtin_amdgcn_tensor_load_to_lds(g0, g1, g2, g3, g4, 0);
}

__device__ __forceinline__ unsigned lds_addr_of(const void* p) {
  return (unsigned)(unsigned long long)p;            // AS(3) offset = low 32 bits
}

// ---------------------------------------------------------------------------
// WMMA helpers (CDNA5 v_wmma_f32_16x16x32_bf16, wave32)
// ---------------------------------------------------------------------------
__device__ __forceinline__ v8f wmma_bf16(v16bf a, v16bf b, v8f c) {
  return __builtin_amdgcn_wmma_f32_16x16x32_bf16(false, a, false, b,
                                                 (short)0, c, false, false);
}

__device__ __forceinline__ v16bf frag_a_lds(const bf16* base, int ld, int lane) {
  const int m = lane & 15, hi = lane >> 4;
  const bf16* p = base + m * ld + hi * 8;
  v8bf lo = *(const v8bf*)p;
  v8bf hh = *(const v8bf*)(p + 16);
  v16bf r;
#pragma unroll
  for (int i = 0; i < 8; ++i) { r[i] = lo[i]; r[i + 8] = hh[i]; }
  return r;
}

__device__ __forceinline__ v16bf frag_b_lds(const bf16* base, int ld, int lane) {
  const int n = lane & 15, hi = lane >> 4;
  const bf16* p = base + n * ld + hi * 8;
  v8bf lo = *(const v8bf*)p;
  v8bf hh = *(const v8bf*)(p + 16);
  v16bf r;
#pragma unroll
  for (int i = 0; i < 8; ++i) { r[i] = lo[i]; r[i + 8] = hh[i]; }
  return r;
}

// ---------------------------------------------------------------------------
// GEMM: C[M,N] = A[M,K](bf16) * W[N,K]^T(bf16) + bias[N]   (einsum bld,od->blo)
// 128x128 block, 8 waves (4x2), wave tile 32x64 = 2x4 wmma. TDM-fed LDS,
// double buffered (wave 0 issues DMA for tile k+1 while all waves compute k).
// ---------------------------------------------------------------------------
__global__ __launch_bounds__(256) void gemm_bf16_nt_kernel(
    const bf16* __restrict__ A, const bf16* __restrict__ W,
    const float* __restrict__ bias, float* __restrict__ C,
    int Mm, int Nn, int Kk) {
  constexpr int BM = 128, BN = 128, BK = 32;
  __shared__ bf16 As[2][BM * BK];
  __shared__ bf16 Bs[2][BM * BK];

  const int tid = threadIdx.x, lane = tid & 31;
  const int wm = (tid >> 5) >> 1, wn = (tid >> 5) & 1;
  const bool issuer = (tid >> 5) == 0;
  const size_t bm = (size_t)blockIdx.x * BM, bn = (size_t)blockIdx.y * BN;

  v8f acc[2][4];
#pragma unroll
  for (int i = 0; i < 2; ++i)
#pragma unroll
    for (int j = 0; j < 4; ++j)
#pragma unroll
      for (int v = 0; v < 8; ++v) acc[i][j][v] = 0.f;

  const bf16* gA = A + bm * (size_t)Kk;
  const bf16* gW = W + bn * (size_t)Kk;
  const int nk = Kk / BK;

  if (issuer) {
    tdm_load_2d(lds_addr_of(&As[0][0]), gA, Kk, Mm, Kk, BK, BM);
    tdm_load_2d(lds_addr_of(&Bs[0][0]), gW, Kk, Nn, Kk, BK, BN);
    __builtin_amdgcn_s_wait_tensorcnt(0);
  }
  __syncthreads();

  int cur = 0;
  for (int kt = 0; kt < nk; ++kt) {
    const int nxt = cur ^ 1;
    const bool more = (kt + 1) < nk;
    if (issuer && more) {
      tdm_load_2d(lds_addr_of(&As[nxt][0]), gA + (size_t)(kt + 1) * BK,
                  Kk, Mm, Kk, BK, BM);
      tdm_load_2d(lds_addr_of(&Bs[nxt][0]), gW + (size_t)(kt + 1) * BK,
                  Kk, Nn, Kk, BK, BN);
    }

    v16bf af[2], bfr[4];
#pragma unroll
    for (int i = 0; i < 2; ++i)
      af[i] = frag_a_lds(&As[cur][(wm * 32 + i * 16) * BK], BK, lane);
#pragma unroll
    for (int j = 0; j < 4; ++j)
      bfr[j] = frag_b_lds(&Bs[cur][(wn * 64 + j * 16) * BK], BK, lane);
#pragma unroll
    for (int i = 0; i < 2; ++i)
#pragma unroll
      for (int j = 0; j < 4; ++j)
        acc[i][j] = wmma_bf16(af[i], bfr[j], acc[i][j]);

    if (issuer && more) __builtin_amdgcn_s_wait_tensorcnt(0);
    __syncthreads();
    cur = nxt;
  }

  const int hi = lane >> 4, nl = lane & 15;
#pragma unroll
  for (int j = 0; j < 4; ++j) {
    const size_t col = bn + wn * 64 + j * 16 + nl;
    const float bv = bias[col];
#pragma unroll
    for (int i = 0; i < 2; ++i)
#pragma unroll
      for (int v = 0; v < 8; ++v) {
        const size_t row = bm + wm * 32 + i * 16 + hi * 8 + v;
        C[row * (size_t)Nn + col] = acc[i][j][v] + bv;
      }
  }
}

// ---------------------------------------------------------------------------
// Fused SiLU-attention per (b,h,q-tile of 64 rows), causal.
// S = (Q K^T)/sqrt(HD); masked -> 0 else silu(S); O = (S@V) * U
// Q/K tiles DMA'd by TDM; V staged transposed (padded stride 72 vs 64 to
// spread LDS banks). LDS = 16+16+18+8 = 58 KB.
// ---------------------------------------------------------------------------
__global__ __launch_bounds__(256) void attn_silu_kernel(
    const bf16* __restrict__ Qb, const bf16* __restrict__ Kb,
    const bf16* __restrict__ Vb, const float* __restrict__ U,
    float* __restrict__ O) {
  constexpr int VP = 72;                            // padded V^T row stride
  __shared__ bf16 qs[64 * 128];
  __shared__ bf16 ks[64 * 128];
  __shared__ bf16 vt[128 * VP];
  __shared__ bf16 st[64 * 64];

  const int tid = threadIdx.x, lane = tid & 31;
  const int wm = (tid >> 5) >> 1, wn = (tid >> 5) & 1;
  const bool issuer = (tid >> 5) == 0;
  const int qt = blockIdx.x, bh = blockIdx.y;
  const int b = bh >> 3, h = bh & 7;
  const int q0 = qt * 64;
  const int r = tid >> 2, c = (tid & 3) * 32;       // V tile-load coordinates

  if (issuer)                                        // DMA Q tile (64x128)
    tdm_load_2d(lds_addr_of(qs), Qb + (size_t)(b * LL + q0) * DD + h * HD,
                HD, LL, DD, HD, 64);

  v8f acco[4];
#pragma unroll
  for (int j = 0; j < 4; ++j)
#pragma unroll
    for (int v = 0; v < 8; ++v) acco[j][v] = 0.f;

  const float scale = 0.08838834764831845f;          // 1/sqrt(128)
  for (int kt = 0; kt <= qt; ++kt) {                 // causal: skip masked tiles
    const int k0 = kt * 64;
    if (issuer)                                      // DMA K tile (64x128)
      tdm_load_2d(lds_addr_of(ks), Kb + (size_t)(b * LL + k0) * DD + h * HD,
                  HD, LL, DD, HD, 64);
    {                                                // V tile, transposed stage
      const bf16* gv = Vb + ((size_t)(b * LL + k0 + r) * DD + h * HD + c);
      v8bf vv[4];
#pragma unroll
      for (int i = 0; i < 4; ++i) vv[i] = *(const v8bf*)(gv + i * 8);
      if (kt < qt) __builtin_prefetch(gv + (size_t)64 * DD, 0, 1);
#pragma unroll
      for (int i = 0; i < 32; ++i) vt[(c + i) * VP + r] = vv[i >> 3][i & 7];
    }
    if (issuer) __builtin_amdgcn_s_wait_tensorcnt(0);
    __syncthreads();

    // ---- S = Q K^T : wave tile 16x32 at (wm*16, wn*32), k over HD=128 ----
    v8f accs[2];
#pragma unroll
    for (int j = 0; j < 2; ++j)
#pragma unroll
      for (int v = 0; v < 8; ++v) accs[j][v] = 0.f;
#pragma unroll
    for (int kk = 0; kk < 128; kk += 32) {
      v16bf aq = frag_a_lds(qs + (wm * 16) * 128 + kk, 128, lane);
#pragma unroll
      for (int j = 0; j < 2; ++j) {
        v16bf bk = frag_b_lds(ks + (wn * 32 + j * 16) * 128 + kk, 128, lane);
        accs[j] = wmma_bf16(aq, bk, accs[j]);
      }
    }
    // scale + causal mask + SiLU -> bf16 S tile
    {
      const int hi = lane >> 4, nl = lane & 15;
#pragma unroll
      for (int j = 0; j < 2; ++j) {
        const int cl = wn * 32 + j * 16 + nl;
        const int cg = k0 + cl;
#pragma unroll
        for (int v = 0; v < 8; ++v) {
          const int rl = wm * 16 + hi * 8 + v;
          const float s = accs[j][v] * scale;
          const float w = (cg <= q0 + rl) ? (s / (1.f + __expf(-s))) : 0.f;
          st[rl * 64 + cl] = (bf16)w;
        }
      }
    }
    __syncthreads();

    // ---- O += S @ V : wave tile 16x64 at (wm*16, wn*64), k over 64 ----
#pragma unroll
    for (int kk = 0; kk < 64; kk += 32) {
      v16bf as = frag_a_lds(st + (wm * 16) * 64 + kk, 64, lane);
#pragma unroll
      for (int j = 0; j < 4; ++j) {
        v16bf bv = frag_b_lds(vt + (wn * 64 + j * 16) * VP + kk, VP, lane);
        acco[j] = wmma_bf16(as, bv, acco[j]);
      }
    }
    __syncthreads();
  }

  // epilogue: multiply by U gate, store fp32 (B,L,D layout)
  const int hi = lane >> 4, nl = lane & 15;
#pragma unroll
  for (int j = 0; j < 4; ++j) {
    const int dcol = wn * 64 + j * 16 + nl;
#pragma unroll
    for (int v = 0; v < 8; ++v) {
      const int rl = wm * 16 + hi * 8 + v;
      const size_t idx = (size_t)(b * LL + q0 + rl) * DD + h * HD + dcol;
      O[idx] = acco[j][v] * U[idx];
    }
  }
}

// ---------------------------------------------------------------------------
// Elementwise / normalization kernels
// ---------------------------------------------------------------------------
__global__ __launch_bounds__(256) void ln_bf16_kernel(
    const float* __restrict__ x, const float* __restrict__ g,
    const float* __restrict__ bta, bf16* __restrict__ out) {
  const int row = blockIdx.x;
  const float* xr = x + (size_t)row * DD;
  __shared__ float s1[256], s2[256];
  float a = 0.f, q = 0.f;
  for (int i = threadIdx.x; i < DD; i += 256) { float v = xr[i]; a += v; q += v * v; }
  s1[threadIdx.x] = a; s2[threadIdx.x] = q;
  __syncthreads();
  for (int s = 128; s > 0; s >>= 1) {
    if (threadIdx.x < (unsigned)s) { s1[threadIdx.x] += s1[threadIdx.x + s]; s2[threadIdx.x] += s2[threadIdx.x + s]; }
    __syncthreads();
  }
  const float m = s1[0] * (1.f / DD);
  const float inv = rsqrtf(s2[0] * (1.f / DD) - m * m + EPS);
  bf16* orow = out + (size_t)row * DD;
  for (int i = threadIdx.x; i < DD; i += 256)
    orow[i] = (bf16)((xr[i] - m) * inv * g[i] + bta[i]);
}

__global__ __launch_bounds__(256) void ln_f32_kernel(
    const float* __restrict__ x, const float* __restrict__ g,
    const float* __restrict__ bta, float* __restrict__ out) {
  const int row = blockIdx.x;
  const float* xr = x + (size_t)row * DD;
  __shared__ float s1[256], s2[256];
  float a = 0.f, q = 0.f;
  for (int i = threadIdx.x; i < DD; i += 256) { float v = xr[i]; a += v; q += v * v; }
  s1[threadIdx.x] = a; s2[threadIdx.x] = q;
  __syncthreads();
  for (int s = 128; s > 0; s >>= 1) {
    if (threadIdx.x < (unsigned)s) { s1[threadIdx.x] += s1[threadIdx.x + s]; s2[threadIdx.x] += s2[threadIdx.x + s]; }
    __syncthreads();
  }
  const float m = s1[0] * (1.f / DD);
  const float inv = rsqrtf(s2[0] * (1.f / DD) - m * m + EPS);
  float* orow = out + (size_t)row * DD;
  for (int i = threadIdx.x; i < DD; i += 256)
    orow[i] = (xr[i] - m) * inv * g[i] + bta[i];
}

__global__ __launch_bounds__(256) void cvt_f32_bf16_kernel(
    const float* __restrict__ s, bf16* __restrict__ d, size_t n) {
  size_t i = (size_t)blockIdx.x * 256 + threadIdx.x;
  if (i < n) d[i] = (bf16)s[i];
}

__global__ __launch_bounds__(256) void rope_bf16_kernel(
    const float* __restrict__ src, bf16* __restrict__ dst) {
  const size_t p = (size_t)blockIdx.x * 256 + threadIdx.x;  // pair index
  if (p >= (size_t)MTOK * (DD / 2)) return;
  const int token = (int)(p / (DD / 2));
  const int w = (int)(p % (DD / 2));
  const int hh = w >> 6;               // head (HD/2 = 64 pairs per head)
  const int i = w & 63;                // freq index
  const int l = token & (LL - 1);      // position
  const float freq = __expf(-(float)(2 * i) * (9.210340371976184f / (float)HD));
  float sn, cs; __sincosf((float)l * freq, &sn, &cs);
  const size_t base = (size_t)token * DD + hh * HD + 2 * i;
  const float x1 = src[base], x2 = src[base + 1];
  dst[base]     = (bf16)(x1 * cs - x2 * sn);
  dst[base + 1] = (bf16)(x1 * sn + x2 * cs);
}

__global__ __launch_bounds__(256) void silu_gate_bf16_kernel(
    const float* __restrict__ p, bf16* __restrict__ o, size_t n) {
  size_t i = (size_t)blockIdx.x * 256 + threadIdx.x;
  if (i < n) { const float v = p[i]; o[i] = (bf16)(v * v / (1.f + __expf(-v))); }
}

__global__ __launch_bounds__(256) void add_inplace_kernel(
    float* __restrict__ x, const float* __restrict__ t, size_t n) {
  size_t i = (size_t)blockIdx.x * 256 + threadIdx.x;
  if (i < n) x[i] += t[i];
}

// ---------------------------------------------------------------------------
// Host orchestration
// ---------------------------------------------------------------------------
extern "C" void kernel_launch(void* const* d_in, const int* in_sizes, int n_in,
                              void* d_out, int out_size, void* d_ws, size_t ws_size,
                              hipStream_t stream) {
  (void)in_sizes; (void)n_in; (void)out_size; (void)ws_size;

  const float* seqs = (const float*)d_in[0];
  const float* Wq = (const float*)d_in[2];  const float* bq = (const float*)d_in[3];
  const float* Wk = (const float*)d_in[4];  const float* bk = (const float*)d_in[5];
  const float* Wu = (const float*)d_in[6];  const float* bu = (const float*)d_in[7];
  const float* Wv = (const float*)d_in[8];  const float* bv = (const float*)d_in[9];
  const float* Wo = (const float*)d_in[10]; const float* bo = (const float*)d_in[11];
  const float* W1 = (const float*)d_in[12]; const float* b1 = (const float*)d_in[13];
  const float* W2 = (const float*)d_in[14]; const float* b2 = (const float*)d_in[15];
  const float* ln1g = (const float*)d_in[16]; const float* ln1b = (const float*)d_in[17];
  const float* ln2g = (const float*)d_in[18]; const float* ln2b = (const float*)d_in[19];
  const float* lnfg = (const float*)d_in[20]; const float* lnfb = (const float*)d_in[21];

  const size_t MD = (size_t)MTOK * DD;          // 4,194,304 elements

  // workspace carving (256B aligned)
  char* base = (char*)d_ws;
  size_t off = 0;
  auto carve = [&](size_t bytes) -> void* {
    void* r = base + off; off += (bytes + 255) & ~(size_t)255; return r;
  };
  float* x   = (float*)carve(MD * 4);             // running activations
  bf16*  hbf = (bf16*)carve(MD * 2);              // bf16 GEMM A-operand buffer
  bf16*  wbf = (bf16*)carve((size_t)DD * DD * 2); // bf16 weight staging
  float* pq  = (float*)carve(MD * 4);             // Q proj / generic GEMM out
  float* pk  = (float*)carve(MD * 4);             // K proj
  float* pu  = (float*)carve(MD * 4);             // U proj (kept fp32)
  float* pv  = (float*)carve(MD * 4);             // V proj / attention out
  bf16*  qb  = (bf16*)carve(MD * 2);
  bf16*  kb  = (bf16*)carve(MD * 2);
  bf16*  vb  = (bf16*)carve(MD * 2);

  const dim3 gemm_grid(MTOK / 128, DD / 128);     // (32, 8)
  const dim3 attn_grid(LL / 64, BB * 8);          // (32, 16)
  const int  ew_grid   = (int)(MD / 256);         // 16384
  const int  pair_grid = (int)(MD / 2 / 256);     // 8192
  const int  w_grid    = (DD * DD) / 256;         // 4096

  auto run_gemm = [&](const bf16* Ain, const float* Wf, const float* biasf,
                      float* Cout) {
    cvt_f32_bf16_kernel<<<w_grid, 256, 0, stream>>>(Wf, wbf, (size_t)DD * DD);
    gemm_bf16_nt_kernel<<<gemm_grid, 256, 0, stream>>>(Ain, wbf, biasf, Cout,
                                                       MTOK, DD, DD);
  };

  // x = seqs
  (void)hipMemcpyAsync(x, seqs, MD * 4, hipMemcpyDeviceToDevice, stream);

  for (int i = 0; i < NLAY; ++i) {
    const size_t wo = (size_t)i * DD * DD, bo_ = (size_t)i * DD;

    // h = LN1(x) in bf16
    ln_bf16_kernel<<<MTOK, 256, 0, stream>>>(x, ln1g + bo_, ln1b + bo_, hbf);

    // Q/K/U/V projections
    run_gemm(hbf, Wq + wo, bq + bo_, pq);
    run_gemm(hbf, Wk + wo, bk + bo_, pk);
    run_gemm(hbf, Wu + wo, bu + bo_, pu);
    run_gemm(hbf, Wv + wo, bv + bo_, pv);

    // RoPE(Q), RoPE(K) -> bf16; V -> bf16
    rope_bf16_kernel<<<pair_grid, 256, 0, stream>>>(pq, qb);
    rope_bf16_kernel<<<pair_grid, 256, 0, stream>>>(pk, kb);
    cvt_f32_bf16_kernel<<<ew_grid, 256, 0, stream>>>(pv, vb, MD);

    // fused causal SiLU-attention, gated by U -> pv (fp32)
    attn_silu_kernel<<<attn_grid, 256, 0, stream>>>(qb, kb, vb, pu, pv);

    // output projection + residual
    cvt_f32_bf16_kernel<<<ew_grid, 256, 0, stream>>>(pv, hbf, MD);
    run_gemm(hbf, Wo + wo, bo + bo_, pq);
    add_inplace_kernel<<<ew_grid, 256, 0, stream>>>(x, pq, MD);

    // MLP: p = LN2(x) W1^T + b1 ; f = (p * silu(p)) W2^T + b2 ; x += f
    ln_bf16_kernel<<<MTOK, 256, 0, stream>>>(x, ln2g + bo_, ln2b + bo_, hbf);
    run_gemm(hbf, W1 + wo, b1 + bo_, pq);
    silu_gate_bf16_kernel<<<ew_grid, 256, 0, stream>>>(pq, hbf, MD);
    run_gemm(hbf, W2 + wo, b2 + bo_, pq);
    add_inplace_kernel<<<ew_grid, 256, 0, stream>>>(x, pq, MD);
  }

  // final LayerNorm -> d_out (fp32)
  ln_f32_kernel<<<MTOK, 256, 0, stream>>>(x, lnfg, lnfb, (float*)d_out);
}
